// ScaledDotProductAttention_91311004713016
// MI455X (gfx1250) — matrix operand order repaired and checked by reading, not compiled, not run on hardware
//
#include <hip/hip_runtime.h>

typedef __attribute__((ext_vector_type(16))) _Float16 v16h;
typedef __attribute__((ext_vector_type(8)))  _Float16 h8;
typedef __attribute__((ext_vector_type(4)))  _Float16 h4;
typedef __attribute__((ext_vector_type(8)))  float    v8f;

namespace {
constexpr int SEQ = 4096;   // sequence length
constexpr int NH  = 16;     // heads
constexpr int HD  = 64;     // head dim
constexpr int BQ  = 64;     // q rows per block (4 waves x 16 rows)
constexpr int BK  = 32;     // k/v rows per tile
constexpr int KP  = 72;     // K-lds row stride in halfs (144B, 16B aligned)
constexpr int VP  = 40;     // V^T-lds row stride in halfs (80B, 16B aligned)
constexpr int PP  = 40;     // P-lds row stride in halfs
// 1/sqrt(64) * log2(e): softmax computed in exp2 domain (v_exp_f32 direct)
constexpr float QSCALE = 0.125f * 1.44269504088896340736f;
}

// LDS offset of a generic pointer into shared memory (for async-to-LDS asm).
__device__ __forceinline__ unsigned lds_off(const void* p) {
  return (unsigned)(size_t)(const __attribute__((address_space(3))) void*)p;
}
// CDNA5 async copy: 16B per lane, global -> LDS, tracked by ASYNCcnt.
__device__ __forceinline__ void async_b128(unsigned ldsoff, const void* g) {
  asm volatile("global_load_async_to_lds_b128 %0, %1, off"
               :: "v"(ldsoff), "v"((unsigned long long)g) : "memory");
}
__device__ __forceinline__ void wait_async0() {
  asm volatile("s_wait_asynccnt 0x0" ::: "memory");
}

// 16-lane row max of max(a,b,m): combine + 4-step DPP butterfly + running max,
// all as raw v_max_num (no canonicalize ops, no mov ties). 6 VALU total.
__device__ __forceinline__ float rowmax16_3(float a, float b, float m) {
  asm("v_max_num_f32_e32 %0, %0, %1\n\t"
      "v_max_num_f32_dpp %0, %0, %0 quad_perm:[1,0,3,2] row_mask:0xf bank_mask:0xf bound_ctrl:1\n\t"
      "v_max_num_f32_dpp %0, %0, %0 quad_perm:[2,3,0,1] row_mask:0xf bank_mask:0xf bound_ctrl:1\n\t"
      "v_max_num_f32_dpp %0, %0, %0 row_half_mirror row_mask:0xf bank_mask:0xf bound_ctrl:1\n\t"
      "v_max_num_f32_dpp %0, %0, %0 row_mirror row_mask:0xf bank_mask:0xf bound_ctrl:1\n\t"
      "v_max_num_f32_e32 %0, %0, %2"
      : "+v"(a) : "v"(b), "v"(m));
  return a;
}

// 16-lane row sum via DPP butterfly; v_add_f32_dpp fuses cleanly from builtin.
template <int CTRL>
__device__ __forceinline__ float dpp_permf(float x) {
  return __int_as_float(
      __builtin_amdgcn_update_dpp(0, __float_as_int(x), CTRL, 0xF, 0xF, true));
}
__device__ __forceinline__ float rowsum16(float x) {
  x += dpp_permf<0x0B1>(x);   // quad_perm xor1
  x += dpp_permf<0x04E>(x);   // quad_perm xor2
  x += dpp_permf<0x141>(x);   // row_half_mirror (xor7)
  x += dpp_permf<0x140>(x);   // row_mirror (xor15)
  return x;
}

// =============== shared per-tile compute body ================================
// One wave = 16 q rows. C/D layout: lane holds column n=lane%16, vgpr r holds
// row m = r + 8*(lane/16). A layout (16x32 f16): lane holds row m=lane%16,
// elems 0..7 -> k = 8*(lane/16)+0..7, elems 8..15 -> +16. B layout (32x16):
// lane holds col n=lane%16, elems 0..7 -> k = 8*(lane/16)+0..7, elems 8..15 -> +16 (as A).
template <bool MASK>
__device__ __forceinline__ void process_tile(
    const _Float16* __restrict__ Kt,   // K tile in LDS   [k][d] stride KP
    const _Float16* __restrict__ Vt,   // V^T tile in LDS [d][k] stride VP
    _Float16* __restrict__ Pw,         // this wave's P transpose buffer
    const v16h (&Qa)[2], v8f (&Oacc)[4], float (&mrow)[8], float (&lrow)[8],
    int k0, int qw, int hh, int l16, int aoff, int boff) {
  const v8f vzero = {};

  // ---- S(16x32) = (Q*scale*log2e) @ K^T : 4x v_wmma_f32_16x16x32_f16
  v8f Sacc[2];
#pragma unroll
  for (int t = 0; t < 2; ++t) {
    const _Float16* bp = &Kt[(t * 16 + l16) * KP + aoff];
    v16h Kb0, Kb1;
#pragma unroll
    for (int e = 0; e < 8; ++e) { Kb0[e] = bp[e]; Kb0[e + 8] = bp[16 + e]; Kb1[e] = bp[32 + e]; Kb1[e + 8] = bp[48 + e]; }
    Sacc[t] = __builtin_amdgcn_wmma_f32_16x16x32_f16(
        false, Qa[0], false, Kb0, (short)0, vzero, false, false);
    Sacc[t] = __builtin_amdgcn_wmma_f32_16x16x32_f16(
        false, Qa[1], false, Kb1, (short)0, Sacc[t], false, false);
  }

  // ---- causal mask (diagonal tiles only) + online softmax (exp2 domain)
#pragma unroll
  for (int r = 0; r < 8; ++r) {
    float s0 = Sacc[0][r], s1 = Sacc[1][r];
    if (MASK) {
      const int row = qw + r + 8 * hh;
      s0 = (k0 + l16      <= row) ? s0 : -__builtin_inff();
      s1 = (k0 + 16 + l16 <= row) ? s1 : -__builtin_inff();
    }
    const float mnew  = rowmax16_3(s0, s1, mrow[r]);
    const float alpha = __builtin_amdgcn_exp2f(mrow[r] - mnew);
    mrow[r] = mnew;

    const float p0 = __builtin_amdgcn_exp2f(s0 - mnew);
    const float p1 = __builtin_amdgcn_exp2f(s1 - mnew);
    Sacc[0][r] = p0;
    Sacc[1][r] = p1;
    lrow[r] = lrow[r] * alpha + rowsum16(p0 + p1);
#pragma unroll
    for (int n = 0; n < 4; ++n) Oacc[n][r] *= alpha;
  }

  // ---- transpose P (C layout -> A layout) through per-wave LDS buffer
#pragma unroll
  for (int t = 0; t < 2; ++t)
#pragma unroll
    for (int r = 0; r < 8; ++r)
      Pw[(r + 8 * hh) * PP + t * 16 + l16] = (_Float16)Sacc[t][r];

  v16h Pa;
  {
    const _Float16* pp = &Pw[l16 * PP + aoff];
#pragma unroll
    for (int e = 0; e < 8; ++e) { Pa[e] = pp[e]; Pa[e + 8] = pp[16 + e]; }
  }

  // ---- O += P @ V : 4x v_wmma_f32_16x16x32_f16 over the D=64 columns
#pragma unroll
  for (int n = 0; n < 4; ++n) {
    const _Float16* vp = &Vt[(n * 16 + l16) * VP + aoff];
    v16h Vb;
#pragma unroll
    for (int e = 0; e < 8; ++e) { Vb[e] = vp[e]; Vb[e + 8] = vp[16 + e]; }
    Oacc[n] = __builtin_amdgcn_wmma_f32_16x16x32_f16(
        false, Pa, false, Vb, (short)0, Oacc[n], false, false);
  }
}

// =============== pre-pass: f32 -> f16 conversion into workspace ==============
__launch_bounds__(256)
__global__ void convert_k16(const float* __restrict__ K, _Float16* __restrict__ K16) {
  const size_t i = ((size_t)blockIdx.x * 256 + threadIdx.x) * 4;
  float4 v = *(const float4*)(K + i);
  h4 o; o[0] = (_Float16)v.x; o[1] = (_Float16)v.y;
        o[2] = (_Float16)v.z; o[3] = (_Float16)v.w;
  *(h4*)(K16 + i) = o;
}

// Transpose V per head: Vt16[h][d][s] = (f16) V[h][s][d], LDS-tiled 64x64.
__launch_bounds__(256)
__global__ void convert_vt16(const float* __restrict__ V, _Float16* __restrict__ Vt16) {
  __shared__ alignas(16) _Float16 tile[64 * 72];
  const int t  = threadIdx.x;
  const int h  = blockIdx.y;
  const int s0 = blockIdx.x * 64;
  const int r  = t >> 2;            // 0..63
  const int cb = (t & 3) * 16;      // 0,16,32,48
  {
    const float4* vp = (const float4*)(V + ((size_t)h * SEQ + s0 + r) * HD + cb);
    float f[16];
#pragma unroll
    for (int q = 0; q < 4; ++q) {
      float4 v4 = vp[q];
      f[q*4+0] = v4.x; f[q*4+1] = v4.y; f[q*4+2] = v4.z; f[q*4+3] = v4.w;
    }
    h8 lo, hi;
#pragma unroll
    for (int e = 0; e < 8; ++e) { lo[e] = (_Float16)f[e]; hi[e] = (_Float16)f[8+e]; }
    *(h8*)&tile[r * 72 + cb]     = lo;
    *(h8*)&tile[r * 72 + cb + 8] = hi;
  }
  __syncthreads();
  {
    h8 lo, hi;                        // out row d = r, s = s0+cb..+15
#pragma unroll
    for (int e = 0; e < 8; ++e) {
      lo[e] = tile[(cb + e)     * 72 + r];
      hi[e] = tile[(cb + 8 + e) * 72 + r];
    }
    _Float16* op = Vt16 + ((size_t)h * HD + r) * SEQ + s0 + cb;
    *(h8*)op       = lo;
    *(h8*)(op + 8) = hi;
  }
}

// =============== main kernel: async-staged flash attention ===================
__launch_bounds__(128)
__global__ void fa_fwd_async(const float* __restrict__ Qg,
                             const _Float16* __restrict__ K16,
                             const _Float16* __restrict__ Vt16,
                             float* __restrict__ Og) {
  __shared__ alignas(16) _Float16 Klds[2][BK * KP];    // K tile [k][d], dbl buf
  __shared__ alignas(16) _Float16 Vtlds[2][HD * VP];   // V tile [d][k], dbl buf
  __shared__ alignas(16) _Float16 Plds[4][16 * PP];    // per-wave P transpose

  const int tid  = threadIdx.x;
  const int wv   = tid >> 5;
  const int lane = tid & 31;
  const int hh   = lane >> 4;
  const int l16  = lane & 15;

  const int head = blockIdx.y;
  const int qb   = blockIdx.x * BQ;
  const int qw   = qb + wv * 16;

  const float*    Qh  = Qg   + (size_t)head * SEQ * HD;
  const _Float16* Kh  = K16  + (size_t)head * SEQ * HD;
  const _Float16* Vth = Vt16 + (size_t)head * HD * SEQ;
  float*          Oh  = Og   + (size_t)head * SEQ * HD;

  const int aoff = hh * 8;
  const int boff = hh * 16;

  // ---- Q rows into WMMA-A regs (two K=32 chunks over D=64), scale folded.
  v16h Qa[2];
  {
    const float* qp = Qh + (size_t)(qw + l16) * HD;
#pragma unroll
    for (int c = 0; c < 2; ++c)
#pragma unroll
      for (int e = 0; e < 8; ++e) {
        Qa[c][e]     = (_Float16)(qp[c * 32 + aoff + e]      * QSCALE);
        Qa[c][e + 8] = (_Float16)(qp[c * 32 + aoff + 16 + e] * QSCALE);
      }
  }

  const v8f vzero = {};
  v8f Oacc[4];
#pragma unroll
  for (int n = 0; n < 4; ++n) Oacc[n] = vzero;
  float mrow[8], lrow[8];
#pragma unroll
  for (int r = 0; r < 8; ++r) { mrow[r] = -__builtin_inff(); lrow[r] = 0.0f; }

  const int nkt = qb / BK + 2;

  // Async staging: per wave 2x K-chunks + 2x Vt-chunks of 32 lanes x 16B.
  auto stage = [&](int kt) {
    const int buf = kt & 1;
    const int k0  = kt * BK;
#pragma unroll
    for (int i = 0; i < 2; ++i) {
      const int c  = wv * 64 + i * 32 + lane;     // 0..255 chunk id
      const int kr = c >> 3, k8 = (c & 7) * 8;    // K tile: [kr][k8..k8+7]
      async_b128(lds_off(&Klds[buf][kr * KP + k8]),
                 Kh + (size_t)(k0 + kr) * HD + k8);
      const int d  = c >> 2, s8 = (c & 3) * 8;    // Vt tile: [d][s8..s8+7]
      async_b128(lds_off(&Vtlds[buf][d * VP + s8]),
                 Vth + (size_t)d * SEQ + k0 + s8);
    }
  };

  stage(0);
  wait_async0();
  __syncthreads();

  int kt = 0;
  for (; kt < nkt - 2; ++kt) {         // interior tiles: no causal masking
    stage(kt + 1);
    process_tile<false>(Klds[kt & 1], Vtlds[kt & 1], Plds[wv],
                        Qa, Oacc, mrow, lrow, kt * BK, qw, hh, l16, aoff, boff);
    wait_async0();
    __syncthreads();
  }
  for (; kt < nkt; ++kt) {             // last two tiles touch the diagonal
    if (kt + 1 < nkt) stage(kt + 1);
    process_tile<true>(Klds[kt & 1], Vtlds[kt & 1], Plds[wv],
                       Qa, Oacc, mrow, lrow, kt * BK, qw, hh, l16, aoff, boff);
    wait_async0();
    __syncthreads();
  }

  // ---- final normalize by row sum and store O (f32)
#pragma unroll
  for (int r = 0; r < 8; ++r) {
    const int row = qw + r + 8 * hh;
    const float inv = __builtin_amdgcn_rcpf(lrow[r]);
#pragma unroll
    for (int n = 0; n < 4; ++n)
      Oh[(size_t)row * HD + n * 16 + l16] = Oacc[n][r] * inv;
  }
}

// =============== fallback: self-contained (no workspace) =====================
__launch_bounds__(128)
__global__ void fa_fwd_standalone(const float* __restrict__ Qg,
                                  const float* __restrict__ Kg,
                                  const float* __restrict__ Vg,
                                  float* __restrict__ Og) {
  __shared__ alignas(16) _Float16 Klds[2][BK * KP];
  __shared__ alignas(16) _Float16 Vtlds[2][HD * VP];
  __shared__ alignas(16) _Float16 Plds[4][16 * PP];

  const int tid  = threadIdx.x;
  const int wv   = tid >> 5;
  const int lane = tid & 31;
  const int hh   = lane >> 4;
  const int l16  = lane & 15;

  const int head = blockIdx.y;
  const int qb   = blockIdx.x * BQ;
  const int qw   = qb + wv * 16;

  const float* Qh = Qg + (size_t)head * SEQ * HD;
  const float* Kh = Kg + (size_t)head * SEQ * HD;
  const float* Vh = Vg + (size_t)head * SEQ * HD;
  float*       Oh = Og + (size_t)head * SEQ * HD;

  const int aoff = hh * 8;
  const int boff = hh * 16;

  v16h Qa[2];
  {
    const float* qp = Qh + (size_t)(qw + l16) * HD;
#pragma unroll
    for (int c = 0; c < 2; ++c)
#pragma unroll
      for (int e = 0; e < 8; ++e) {
        Qa[c][e]     = (_Float16)(qp[c * 32 + aoff + e]      * QSCALE);
        Qa[c][e + 8] = (_Float16)(qp[c * 32 + aoff + 16 + e] * QSCALE);
      }
  }

  const v8f vzero = {};
  v8f Oacc[4];
#pragma unroll
  for (int n = 0; n < 4; ++n) Oacc[n] = vzero;
  float mrow[8], lrow[8];
#pragma unroll
  for (int r = 0; r < 8; ++r) { mrow[r] = -__builtin_inff(); lrow[r] = 0.0f; }

  const int nkt   = qb / BK + 2;
  const int ksrow = tid >> 2;
  const int kscol = (tid & 3) * 16;
  const int vrb   = (tid >> 4) * 4;
  const int vcb   = (tid & 15) * 4;

  auto stage = [&](int kt) {
    const int buf = kt & 1;
    const int k0  = kt * BK;
    {
      const float4* kp4 = (const float4*)(Kh + (size_t)(k0 + ksrow) * HD + kscol);
      float kf[16];
#pragma unroll
      for (int q = 0; q < 4; ++q) {
        float4 k4 = kp4[q];
        kf[q*4+0] = k4.x; kf[q*4+1] = k4.y; kf[q*4+2] = k4.z; kf[q*4+3] = k4.w;
      }
      h8 lo, hi;
#pragma unroll
      for (int e = 0; e < 8; ++e) { lo[e] = (_Float16)kf[e]; hi[e] = (_Float16)kf[8+e]; }
      *(h8*)&Klds[buf][ksrow * KP + kscol]     = lo;
      *(h8*)&Klds[buf][ksrow * KP + kscol + 8] = hi;
    }
    {
      float vf[4][4];
#pragma unroll
      for (int i = 0; i < 4; ++i) {
        float4 v4 = *(const float4*)(Vh + (size_t)(k0 + vrb + i) * HD + vcb);
        vf[i][0] = v4.x; vf[i][1] = v4.y; vf[i][2] = v4.z; vf[i][3] = v4.w;
      }
#pragma unroll
      for (int j = 0; j < 4; ++j) {
        h4 col;
#pragma unroll
        for (int i = 0; i < 4; ++i) col[i] = (_Float16)vf[i][j];
        *(h4*)&Vtlds[buf][(vcb + j) * VP + vrb] = col;
      }
    }
  };

  stage(0);
  __syncthreads();

  int kt = 0;
  for (; kt < nkt - 2; ++kt) {
    stage(kt + 1);
    process_tile<false>(Klds[kt & 1], Vtlds[kt & 1], Plds[wv],
                        Qa, Oacc, mrow, lrow, kt * BK, qw, hh, l16, aoff, boff);
    __syncthreads();
  }
  for (; kt < nkt; ++kt) {
    if (kt + 1 < nkt) stage(kt + 1);
    process_tile<true>(Klds[kt & 1], Vtlds[kt & 1], Plds[wv],
                       Qa, Oacc, mrow, lrow, kt * BK, qw, hh, l16, aoff, boff);
    __syncthreads();
  }

#pragma unroll
  for (int r = 0; r < 8; ++r) {
    const int row = qw + r + 8 * hh;
    const float inv = __builtin_amdgcn_rcpf(lrow[r]);
#pragma unroll
    for (int n = 0; n < 4; ++n)
      Oh[(size_t)row * HD + n * 16 + l16] = Oacc[n][r] * inv;
  }
}

extern "C" void kernel_launch(void* const* d_in, const int* in_sizes, int n_in,
                              void* d_out, int out_size, void* d_ws, size_t ws_size,
                              hipStream_t stream) {
  (void)in_sizes; (void)n_in; (void)out_size;
  const float* Q = (const float*)d_in[0];
  const float* K = (const float*)d_in[1];
  const float* V = (const float*)d_in[2];
  float* O = (float*)d_out;

  const size_t nelem = (size_t)NH * SEQ * HD;          // per-tensor elements
  const size_t need  = nelem * 2 * sizeof(_Float16);   // K16 + Vt16 = 16 MB

  if (ws_size >= need) {
    _Float16* K16  = (_Float16*)d_ws;
    _Float16* Vt16 = (_Float16*)((char*)d_ws + nelem * sizeof(_Float16));
    convert_k16<<<dim3((unsigned)(nelem / 1024)), 256, 0, stream>>>(K, K16);
    convert_vt16<<<dim3(SEQ / 64, NH), 256, 0, stream>>>(V, Vt16);
    fa_fwd_async<<<dim3(SEQ / BQ, NH), 128, 0, stream>>>(Q, K16, Vt16, O);
  } else {
    fa_fwd_standalone<<<dim3(SEQ / BQ, NH), 128, 0, stream>>>(Q, K, V, O);
  }
}
